// CNN_xLSTM_AMT_81527069213428
// MI455X (gfx1250) — matrix operand order repaired and checked by reading, not compiled, not run on hardware
//
#include <hip/hip_runtime.h>
#include <hip/hip_bf16.h>

// ---------------------------------------------------------------------------
// CDNA5 (gfx1250) implementation: CNN + xLSTM-attention + classifier.
// Dense contractions -> v_wmma_f32_16x16x32_f16 (wave32, fp32 accum).
// Row-major-B GEMM / conv-weight slabs are staged into LDS with the Tensor
// Data Mover, double-buffered so the DMA overlaps the WMMA pipe:
//   issue slab i+1 -> s_wait_tensorcnt 1 (slab i landed) -> barrier -> compute.
// ---------------------------------------------------------------------------

typedef __attribute__((ext_vector_type(16))) _Float16 v16h;
typedef __attribute__((ext_vector_type(8)))  float    v8f;

#define ELU_F(y) ((y) > 0.f ? (y) : (__expf(y) - 1.f))

#ifndef USE_TDM
#define USE_TDM 1
#endif
#if USE_TDM && defined(__has_builtin)
#if __has_builtin(__builtin_amdgcn_tensor_load_to_lds) && \
    __has_builtin(__builtin_amdgcn_s_wait_tensorcnt)
#define HAVE_TDM 1
#endif
#endif
#ifndef HAVE_TDM
#define HAVE_TDM 0
#endif

#if HAVE_TDM
typedef __attribute__((ext_vector_type(4))) unsigned u32x4;
typedef __attribute__((ext_vector_type(4))) int      i32x4;
typedef __attribute__((ext_vector_type(8))) int      i32x8;

// Issue (no wait) a 2-D f16 tile DMA: global (row-major, stride_elems) -> LDS.
// D# packing per CDNA5 ISA 8.3/8.4: count=1, data_size=1 (2 bytes),
// tensor_dim = remaining extent (OOB reads -> zero-fill), type=2.
__device__ __forceinline__ void tdm_issue_2d_f16(void* ldsdst, const void* gsrc,
                                                 unsigned tile_cols, unsigned tile_rows,
                                                 unsigned stride_elems,
                                                 unsigned rem_cols, unsigned rem_rows) {
  unsigned long long ga = (unsigned long long)(uintptr_t)gsrc;
  u32x4 g0;
  g0.x = 1u;                                   // count=1 (valid descriptor)
  g0.y = (unsigned)(uintptr_t)ldsdst;          // lds_addr (bytes)
  g0.z = (unsigned)ga;                         // global_addr[31:0]
  g0.w = ((unsigned)(ga >> 32) & 0x1FFFFFFu) | (2u << 30);  // ga[56:32] | type=2
  i32x8 g1;
  g1[0] = (int)(1u << 16);                                       // data_size=2B
  g1[1] = (int)((rem_cols & 0xFFFFu) << 16);                     // tensor_dim0 lo
  g1[2] = (int)(((rem_cols >> 16) & 0xFFFFu) |                   // tensor_dim0 hi
                ((rem_rows & 0xFFFFu) << 16));                   // tensor_dim1 lo
  g1[3] = (int)(((rem_rows >> 16) & 0xFFFFu) |                   // tensor_dim1 hi
                ((tile_cols & 0xFFFFu) << 16));                  // tile_dim0
  g1[4] = (int)(tile_rows & 0xFFFFu);                            // tile_dim1
  g1[5] = (int)stride_elems;                                     // dim0_stride lo
  g1[6] = 0;
  g1[7] = 0;
  i32x4 z4 = {0, 0, 0, 0};
#if __clang_major__ >= 23
  i32x8 z8 = {0, 0, 0, 0, 0, 0, 0, 0};
  __builtin_amdgcn_tensor_load_to_lds(g0, g1, z4, z4, z8, 0);
#else
  __builtin_amdgcn_tensor_load_to_lds(g0, g1, z4, z4, 0);
#endif
}
#endif  // HAVE_TDM

// ---------------------------------------------------------------------------
// NN GEMM:  C[M,N] = alpha * A[M,K] @ B[K,N] (+bias +resid, opt ELU)
// Block = 128 threads (4 waves) -> 64(M) x 64(N) macro-tile.
//   wave w owns N-subtile w; 4 accumulators walk 4 M-subtiles (B-frag reuse x4).
//   B slab [32 x 64] double-buffered in LDS, filled by TDM (pipelined) or
//   cooperative copy; shared by all 4 waves.
// Requires M % 64 == 0, K % 32 == 0. Columns masked against N.
// flags&2 -> ELU.
// ---------------------------------------------------------------------------
__global__ void gemm_nn_wmma(const _Float16* __restrict__ A,
                             const _Float16* __restrict__ Bm,
                             const float* __restrict__ bias,
                             const float* __restrict__ resid,
                             float* __restrict__ Cf,
                             _Float16* __restrict__ Ch,
                             int N, int K, int lda, int ldb, int ldc,
                             long long sA, long long sB, long long sC,
                             float alpha, int flags) {
  __shared__ _Float16 ldsB[2][32 * 64];
  const int lane = threadIdx.x & 31;
  const int wave = threadIdx.x >> 5;
  const int z = blockIdx.z;
  A  += (long long)z * sA;
  Bm += (long long)z * sB;
  const long long cbase = (long long)z * sC;

  const int c0 = blockIdx.x * 64;
  const int cl = wave * 16 + (lane & 15);  // column within macro-tile
  const int c  = c0 + cl;
  const int rowblk = blockIdx.y * 64;
  const int kbA = (lane < 16) ? 0 : 8;
  const int kbB = (lane < 16) ? 0 : 8;
  const int nsteps = K / 32;

#if HAVE_TDM
  if (wave == 0)
    tdm_issue_2d_f16(ldsB[0], Bm + c0, 64u, 32u,
                     (unsigned)ldb, (unsigned)(N - c0), (unsigned)K);
#endif

  v8f acc[4] = {};
  for (int step = 0; step < nsteps; ++step) {
    const int k0 = step * 32;
#if HAVE_TDM
    if (wave == 0) {
      if (step + 1 < nsteps) {
        tdm_issue_2d_f16(ldsB[(step + 1) & 1],
                         Bm + (long long)(k0 + 32) * ldb + c0, 64u, 32u,
                         (unsigned)ldb, (unsigned)(N - c0),
                         (unsigned)(K - k0 - 32));
        __builtin_amdgcn_s_wait_tensorcnt(1);  // slab `step` has landed
      } else {
        __builtin_amdgcn_s_wait_tensorcnt(0);
      }
    }
#else
    for (int idx = threadIdx.x; idx < 32 * 64; idx += (int)blockDim.x) {
      int kr = idx >> 6, cc = idx & 63;
      _Float16 v = (_Float16)0.f;
      if (c0 + cc < N) v = Bm[(long long)(k0 + kr) * ldb + (c0 + cc)];
      ldsB[step & 1][idx] = v;
    }
#endif
    __syncthreads();

    const _Float16* bs = ldsB[step & 1];
    v16h b;
#pragma unroll
    for (int e = 0; e < 16; ++e) b[e] = bs[(kbB + e + (e & 8)) * 64 + cl];

#pragma unroll
    for (int sub = 0; sub < 4; ++sub) {
      const int r = rowblk + sub * 16 + (lane & 15);
      v16h a;
#pragma unroll
      for (int e = 0; e < 16; ++e) {
        int v  = e >> 1, hh = e & 1;
        int kk = (v < 4) ? (kbA + 2 * v + hh) : (16 + kbA + 2 * (v - 4) + hh);
        a[e] = A[(long long)r * lda + k0 + kk];
      }
      acc[sub] = __builtin_amdgcn_wmma_f32_16x16x32_f16(false, a, false, b,
                                                        (short)0, acc[sub],
                                                        false, false);
    }
    __syncthreads();  // all reads of this buffer done before it is re-filled
  }

  const int rb = (lane < 16) ? 0 : 8;
  if (c < N) {
    const float bv = bias ? bias[c] : 0.f;
#pragma unroll
    for (int sub = 0; sub < 4; ++sub) {
#pragma unroll
      for (int e = 0; e < 8; ++e) {
        long long idx = cbase + (long long)(rowblk + sub * 16 + rb + e) * ldc + c;
        float y = acc[sub][e] * alpha + bv;
        if (resid) y += resid[idx];
        if (flags & 2) y = ELU_F(y);
        if (Cf) Cf[idx] = y;
        if (Ch) Ch[idx] = (_Float16)y;
      }
    }
  }
}

// ---------------------------------------------------------------------------
// TN GEMM (B stored [N,K], B^T access): used for attn = q k^T. One 16x16
// tile per wave; both fragments are contiguous (b128-vectorizable).
// ---------------------------------------------------------------------------
__global__ void gemm_tn_wmma(const _Float16* __restrict__ A,
                             const _Float16* __restrict__ Bt,
                             float* __restrict__ Cf,
                             int N, int K, int lda, int ldb, int ldc,
                             long long sA, long long sB, long long sC,
                             float alpha) {
  const int lane = threadIdx.x & 31;
  const int wave = threadIdx.x >> 5;
  const int tn = blockIdx.x * 4 + wave;
  if (tn * 16 >= N) return;
  const int tm = blockIdx.y;
  const int z  = blockIdx.z;
  A  += (long long)z * sA;
  Bt += (long long)z * sB;
  const long long cbase = (long long)z * sC;

  const int r = tm * 16 + (lane & 15);
  const int c = tn * 16 + (lane & 15);
  const int kbA = (lane < 16) ? 0 : 8;
  const int kbB = (lane < 16) ? 0 : 8;

  v8f acc = {};
  for (int k0 = 0; k0 < K; k0 += 32) {
    v16h a, b;
#pragma unroll
    for (int e = 0; e < 16; ++e) {
      int v  = e >> 1, hh = e & 1;
      int kk = (v < 4) ? (kbA + 2 * v + hh) : (16 + kbA + 2 * (v - 4) + hh);
      a[e] = A[(long long)r * lda + k0 + kk];
    }
#pragma unroll
    for (int e = 0; e < 16; ++e)
      b[e] = Bt[(long long)c * ldb + k0 + kbB + e + (e & 8)];
    acc = __builtin_amdgcn_wmma_f32_16x16x32_f16(false, a, false, b,
                                                 (short)0, acc, false, false);
  }
  const int rb = (lane < 16) ? 0 : 8;
#pragma unroll
  for (int e = 0; e < 8; ++e) {
    long long idx = cbase + (long long)(tm * 16 + rb + e) * ldc + c;
    Cf[idx] = acc[e] * alpha;
  }
}

// ---------------------------------------------------------------------------
// Implicit-GEMM conv 3x3 SAME + BN + ELU via WMMA.
//   in : f16 [B][Cin][F][T], W: f16 [Cout][Cin*9], out: f16 [B][Cout][F][T]
//   Block = 4 waves; each wave owns one 16-wide spatial tile and walks 4
//   Cout-subtiles (64 channels) -> im2col gather built ONCE per K-step and
//   reused by 4 WMMAs. Weight slab [64 x 32] double-buffered in LDS,
//   TDM-pipelined (or cooperative copy fallback).
//   Requires (F*T/16) % 4 == 0 and Cout % 64 == 0 (true for all layers).
// ---------------------------------------------------------------------------
__global__ void conv_wmma(const _Float16* __restrict__ in,
                          const _Float16* __restrict__ Wf,
                          const float* __restrict__ g, const float* __restrict__ bb,
                          const float* __restrict__ mm, const float* __restrict__ vv,
                          _Float16* __restrict__ out,
                          int Cin, int Cout, int F, int T) {
  __shared__ _Float16 ldsA[2][64 * 32];
  const int lane = threadIdx.x & 31;
  const int wave = threadIdx.x >> 5;
  const int NSP  = F * T;
  const int tn   = blockIdx.x * 4 + wave;  // spatial tile (exact fit)
  const int cobase = blockIdx.y * 64;
  const int n    = blockIdx.z;
  const int K    = Cin * 9;

  const int s = tn * 16 + (lane & 15);
  const int f = s / T;
  const int t = s - f * T;
  const int kbA = (lane < 16) ? 0 : 8;
  const int kbB = (lane < 16) ? 0 : 8;
  const int nsteps = K / 32;

  const _Float16* inb = in + (long long)n * Cin * NSP;

#if HAVE_TDM
  if (wave == 0)
    tdm_issue_2d_f16(ldsA[0], Wf + (long long)cobase * K, 32u, 64u,
                     (unsigned)K, (unsigned)K, (unsigned)(Cout - cobase));
#endif

  v8f acc[4] = {};
  for (int step = 0; step < nsteps; ++step) {
    const int k0 = step * 32;
#if HAVE_TDM
    if (wave == 0) {
      if (step + 1 < nsteps) {
        tdm_issue_2d_f16(ldsA[(step + 1) & 1],
                         Wf + (long long)cobase * K + k0 + 32, 32u, 64u,
                         (unsigned)K, (unsigned)(K - k0 - 32),
                         (unsigned)(Cout - cobase));
        __builtin_amdgcn_s_wait_tensorcnt(1);  // slab `step` has landed
      } else {
        __builtin_amdgcn_s_wait_tensorcnt(0);
      }
    }
#else
    for (int idx = threadIdx.x; idx < 64 * 32; idx += (int)blockDim.x) {
      int r = idx >> 5, cc = idx & 31;
      ldsA[step & 1][idx] = Wf[(long long)(cobase + r) * K + k0 + cc];
    }
#endif
    __syncthreads();

    // im2col gather: one B fragment per wave per K-step (reused by 4 WMMAs).
    v16h b;
#pragma unroll
    for (int e = 0; e < 16; ++e) {
      int kk = k0 + kbB + e + (e & 8);
      int ci = kk / 9;
      int r9 = kk - ci * 9;
      int d3 = r9 / 3;
      int fi = f + d3 - 1;
      int ti = t + (r9 - d3 * 3) - 1;
      _Float16 xv = (_Float16)0.f;
      if (fi >= 0 && fi < F && ti >= 0 && ti < T)
        xv = inb[((long long)ci * F + fi) * T + ti];
      b[e] = xv;
    }

    const _Float16* as = ldsA[step & 1];
#pragma unroll
    for (int sub = 0; sub < 4; ++sub) {
      const int rl = sub * 16 + (lane & 15);
      v16h a;
#pragma unroll
      for (int e = 0; e < 16; ++e) {
        int v  = e >> 1, hh = e & 1;
        int kk = (v < 4) ? (kbA + 2 * v + hh) : (16 + kbA + 2 * (v - 4) + hh);
        a[e] = as[rl * 32 + kk];
      }
      acc[sub] = __builtin_amdgcn_wmma_f32_16x16x32_f16(false, a, false, b,
                                                        (short)0, acc[sub],
                                                        false, false);
    }
    __syncthreads();  // all reads of this buffer done before it is re-filled
  }

  const int rb = (lane < 16) ? 0 : 8;
  _Float16* outb = out + (long long)n * Cout * NSP;
#pragma unroll
  for (int sub = 0; sub < 4; ++sub) {
#pragma unroll
    for (int e = 0; e < 8; ++e) {
      int   co = cobase + sub * 16 + rb + e;
      float sc = g[co] * rsqrtf(vv[co] + 1e-5f);
      float y  = acc[sub][e] * sc + (bb[co] - mm[co] * sc);
      y = ELU_F(y);
      outb[(long long)co * NSP + s] = (_Float16)y;
    }
  }
}

// ---------------------------------------------------------------------------
// conv1 (Cin=1, Cout=32): direct, tiny share of FLOPs. x fp32 -> f16 out.
// ---------------------------------------------------------------------------
__global__ void conv1_direct(const float* __restrict__ x, const float* __restrict__ w,
                             const float* __restrict__ g, const float* __restrict__ bb,
                             const float* __restrict__ mm, const float* __restrict__ vv,
                             _Float16* __restrict__ out, int F, int T) {
  long long i = (long long)blockIdx.x * blockDim.x + threadIdx.x;
  const int CO = 32;
  long long total = (long long)8 * CO * F * T;
  if (i >= total) return;
  int t = (int)(i % T); long long rr = i / T;
  int f = (int)(rr % F); rr /= F;
  int co = (int)(rr % CO);
  int n  = (int)(rr / CO);
  float acc = 0.f;
#pragma unroll
  for (int df = 0; df < 3; ++df)
#pragma unroll
    for (int dt = 0; dt < 3; ++dt) {
      int fi = f + df - 1, ti = t + dt - 1;
      if (fi >= 0 && fi < F && ti >= 0 && ti < T)
        acc += w[co * 9 + df * 3 + dt] * x[((long long)n * F + fi) * T + ti];
    }
  float sc = g[co] * rsqrtf(vv[co] + 1e-5f);
  float y  = acc * sc + (bb[co] - mm[co] * sc);
  y = ELU_F(y);
  out[i] = (_Float16)y;
}

// maxpool (2,1) over frequency axis, VALID.
__global__ void maxpool_f2(const _Float16* __restrict__ in, _Float16* __restrict__ out,
                           int C, int Fin, int Fout, int T) {
  long long i = (long long)blockIdx.x * blockDim.x + threadIdx.x;
  long long total = (long long)8 * C * Fout * T;
  if (i >= total) return;
  int t = (int)(i % T); long long rr = i / T;
  int fo = (int)(rr % Fout); rr /= Fout;
  int c  = (int)(rr % C);
  int n  = (int)(rr / C);
  const _Float16* p = in + (((long long)n * C + c) * Fin + 2 * fo) * T + t;
  _Float16 a = p[0], b = p[T];
  out[i] = (a > b) ? a : b;
}

// [8][C][F][T] -> [8][T][C*F]
__global__ void transpose_cft(const _Float16* __restrict__ in, _Float16* __restrict__ out,
                              int C, int F, int T) {
  long long i = (long long)blockIdx.x * blockDim.x + threadIdx.x;
  long long total = (long long)8 * T * C * F;
  if (i >= total) return;
  int cf = (int)(i % (C * F)); long long rr = i / (C * F);
  int t = (int)(rr % T);
  int n = (int)(rr / T);
  int c = cf / F, f = cf - c * F;
  out[i] = in[(((long long)n * C + c) * F + f) * T + t];
}

// LayerNorm over D; one block per row. Optional +sinusoidal PE, f32/f16 outs.
__global__ void layernorm(const float* __restrict__ in,
                          const float* __restrict__ gamma, const float* __restrict__ beta,
                          float* __restrict__ outf, _Float16* __restrict__ outh,
                          int D, int addpe) {
  const int row = blockIdx.x;
  const int tid = threadIdx.x;
  const float* xr = in + (long long)row * D;
  __shared__ float s1[256], s2[256];
  float a = 0.f, b = 0.f;
  for (int d = tid; d < D; d += blockDim.x) { float v = xr[d]; a += v; b += v * v; }
  s1[tid] = a; s2[tid] = b; __syncthreads();
  for (int st = 128; st > 0; st >>= 1) {
    if (tid < st) { s1[tid] += s1[tid + st]; s2[tid] += s2[tid + st]; }
    __syncthreads();
  }
  const float mu   = s1[0] / (float)D;
  const float var  = s2[0] / (float)D - mu * mu;
  const float rinv = rsqrtf(var + 1e-5f);
  const int   t    = row & 1023;  // T = 1024
  for (int d = tid; d < D; d += blockDim.x) {
    float y = (xr[d] - mu) * rinv * gamma[d] + beta[d];
    if (addpe) {
      int   ii  = d >> 1;
      float div = __expf((float)(2 * ii) * (-9.210340371976184f / (float)D));
      y += (d & 1) ? __cosf((float)t * div) : __sinf((float)t * div);
    }
    if (outf) outf[(long long)row * D + d] = y;
    if (outh) outh[(long long)row * D + d] = (_Float16)y;
  }
}

// i_g[row] = exp(min(z_row . wi + bi, 8))
__global__ void gate_kernel(const _Float16* __restrict__ z, const float* __restrict__ wi,
                            const float* __restrict__ bi, float* __restrict__ gate, int D) {
  const int row = blockIdx.x;
  const int tid = threadIdx.x;
  __shared__ float sb[128];
  float s = 0.f;
  for (int d = tid; d < D; d += blockDim.x)
    s += (float)z[(long long)row * D + d] * wi[d];
  sb[tid] = s; __syncthreads();
  for (int st = 64; st > 0; st >>= 1) {
    if (tid < st) sb[tid] += sb[tid + st];
    __syncthreads();
  }
  if (tid == 0) gate[row] = __expf(fminf(sb[0] + bi[0], 8.0f));
}

// Causal softmax over key axis, times per-key gate -> f16. grid (T, B).
__global__ void softmax_gate(const float* __restrict__ attn, const float* __restrict__ gate,
                             _Float16* __restrict__ gated, int T) {
  const int qi = blockIdx.x;
  const int b  = blockIdx.y;
  const int tid = threadIdx.x;
  const float* arow = attn + ((long long)b * T + qi) * T;
  __shared__ float sb[256];
  __shared__ float smax, ssum;
  float mx = -3.0e38f;
  for (int j = tid; j <= qi; j += blockDim.x) mx = fmaxf(mx, arow[j]);
  sb[tid] = mx; __syncthreads();
  for (int st = 128; st > 0; st >>= 1) {
    if (tid < st) sb[tid] = fmaxf(sb[tid], sb[tid + st]);
    __syncthreads();
  }
  if (tid == 0) smax = sb[0];
  __syncthreads();
  float se = 0.f;
  for (int j = tid; j <= qi; j += blockDim.x) se += __expf(arow[j] - smax);
  sb[tid] = se; __syncthreads();
  for (int st = 128; st > 0; st >>= 1) {
    if (tid < st) sb[tid] += sb[tid + st];
    __syncthreads();
  }
  if (tid == 0) ssum = sb[0];
  __syncthreads();
  const float inv = 1.f / ssum;
  _Float16* grow = gated + ((long long)b * T + qi) * T;
  for (int j = tid; j < T; j += blockDim.x) {
    float p = (j <= qi) ? __expf(arow[j] - smax) * inv : 0.f;
    grow[j] = (_Float16)(p * gate[(long long)b * T + j]);
  }
}

// fp32 -> f16 elementwise (weight packing).
__global__ void cvt_f16(const float* __restrict__ src, _Float16* __restrict__ dst, long long n) {
  long long i = (long long)blockIdx.x * blockDim.x + threadIdx.x;
  if (i < n) dst[i] = (_Float16)src[i];
}

// Pad cls_w2 [256,88] -> f16 [256,96]; cls_b2 [88] -> f32 [96].
__global__ void pad_w2(const float* __restrict__ w, const float* __restrict__ b2,
                       _Float16* __restrict__ wp, float* __restrict__ bp) {
  int i = blockIdx.x * blockDim.x + threadIdx.x;
  if (i < 256 * 96) {
    int k = i / 96, nn = i - k * 96;
    wp[i] = (_Float16)((nn < 88) ? w[k * 88 + nn] : 0.f);
  }
  if (i < 96) bp[i] = (i < 88) ? b2[i] : 0.f;
}

// [(8*T)][96] fp32 -> out [8][88][T] fp32 (== (8,1,88,1024)).
__global__ void write_out(const float* __restrict__ tmp, float* __restrict__ out,
                          int T, int NOUT, int NP) {
  long long i = (long long)blockIdx.x * blockDim.x + threadIdx.x;
  long long total = (long long)8 * NOUT * T;
  if (i >= total) return;
  int t = (int)(i % T); long long rr = i / T;
  int no = (int)(rr % NOUT);
  int b  = (int)(rr / NOUT);
  out[i] = tmp[((long long)b * T + t) * NP + no];
}

// ---------------------------------------------------------------------------
// Host orchestration.
// ---------------------------------------------------------------------------
extern "C" void kernel_launch(void* const* d_in, const int* in_sizes, int n_in,
                              void* d_out, int out_size, void* d_ws, size_t ws_size,
                              hipStream_t stream) {
  (void)in_sizes; (void)n_in; (void)out_size; (void)ws_size;

  const int B = 8, T = 1024, D = 512, HD = 64, L = 4;
  const int F0 = 229, F1 = 114, F2 = 57, F3 = 28;
  const int CNN_OUT = 256 * F3;  // 7168
  const int M = B * T;           // 8192 rows

  // ---- inputs (setup_inputs order) ----
  const float* x       = (const float*)d_in[0];
  const float* conv_w[4] = {(const float*)d_in[1], (const float*)d_in[6],
                            (const float*)d_in[11], (const float*)d_in[16]};
  const float* bn_g[4] = {(const float*)d_in[2], (const float*)d_in[7],
                          (const float*)d_in[12], (const float*)d_in[17]};
  const float* bn_b[4] = {(const float*)d_in[3], (const float*)d_in[8],
                          (const float*)d_in[13], (const float*)d_in[18]};
  const float* bn_m[4] = {(const float*)d_in[4], (const float*)d_in[9],
                          (const float*)d_in[14], (const float*)d_in[19]};
  const float* bn_v[4] = {(const float*)d_in[5], (const float*)d_in[10],
                          (const float*)d_in[15], (const float*)d_in[20]};
  const float* proj_w = (const float*)d_in[21];
  const float* proj_b = (const float*)d_in[22];
  const float* ln0_g  = (const float*)d_in[23];
  const float* ln0_b  = (const float*)d_in[24];
  const float* wq = (const float*)d_in[25];  const float* bq = (const float*)d_in[26];
  const float* wk = (const float*)d_in[27];  const float* bk = (const float*)d_in[28];
  const float* wv = (const float*)d_in[29];  const float* bv = (const float*)d_in[30];
  const float* wi = (const float*)d_in[31];  const float* bi = (const float*)d_in[32];
  const float* wo = (const float*)d_in[33];  const float* bo = (const float*)d_in[34];
  const float* lng = (const float*)d_in[35]; const float* lnb = (const float*)d_in[36];
  const float* cls_lng = (const float*)d_in[37]; const float* cls_lnb = (const float*)d_in[38];
  const float* cls_w1 = (const float*)d_in[39];  const float* cls_b1 = (const float*)d_in[40];
  const float* cls_w2 = (const float*)d_in[41];  const float* cls_b2 = (const float*)d_in[42];

  // ---- workspace carve-out ----
  size_t off = 0;
  auto alloc = [&](size_t nbytes) -> void* {
    void* p = (char*)d_ws + off;
    off += (nbytes + 255) & ~(size_t)255;
    return p;
  };
  _Float16* P  = (_Float16*)alloc((size_t)8 * 32 * F0 * T * 2);   // 120 MB ping
  _Float16* Q  = (_Float16*)alloc((size_t)8 * 64 * F0 * T * 2);   // 240 MB pong
  _Float16* w2f = (_Float16*)alloc((size_t)64 * 32 * 9 * 2);
  _Float16* w3f = (_Float16*)alloc((size_t)128 * 64 * 9 * 2);
  _Float16* w4f = (_Float16*)alloc((size_t)256 * 128 * 9 * 2);
  _Float16* projw16 = (_Float16*)alloc((size_t)CNN_OUT * D * 2);
  _Float16* wq16 = (_Float16*)alloc((size_t)L * D * HD * 2);
  _Float16* wk16 = (_Float16*)alloc((size_t)L * D * HD * 2);
  _Float16* wv16 = (_Float16*)alloc((size_t)L * D * D * 2);
  _Float16* wo16 = (_Float16*)alloc((size_t)L * D * D * 2);
  _Float16* w1c16 = (_Float16*)alloc((size_t)D * 256 * 2);
  _Float16* w2p16 = (_Float16*)alloc((size_t)256 * 96 * 2);
  float*    b2p   = (float*)alloc((size_t)96 * 4);
  float*    h     = (float*)alloc((size_t)M * D * 4);
  _Float16* z16   = (_Float16*)alloc((size_t)M * D * 2);
  _Float16* q16   = (_Float16*)alloc((size_t)M * HD * 2);
  _Float16* k16   = (_Float16*)alloc((size_t)M * HD * 2);
  _Float16* v16   = (_Float16*)alloc((size_t)M * D * 2);
  float*    gate  = (float*)alloc((size_t)M * 4);
  float*    attn  = (float*)alloc((size_t)B * T * T * 4);
  _Float16* gated = (_Float16*)alloc((size_t)B * T * T * 2);
  _Float16* o16   = (_Float16*)alloc((size_t)M * D * 2);
  _Float16* hcls16 = (_Float16*)alloc((size_t)M * D * 2);
  _Float16* c1_16  = (_Float16*)alloc((size_t)M * 256 * 2);
  float*    outp   = (float*)alloc((size_t)M * 96 * 4);

  auto cvt = [&](const float* s, _Float16* d, long long n) {
    cvt_f16<<<dim3((unsigned)((n + 255) / 256)), dim3(256), 0, stream>>>(s, d, n);
  };

  // ---- weight packing (f32 -> f16) ----
  cvt(conv_w[1], w2f, (long long)64 * 32 * 9);
  cvt(conv_w[2], w3f, (long long)128 * 64 * 9);
  cvt(conv_w[3], w4f, (long long)256 * 128 * 9);
  cvt(proj_w, projw16, (long long)CNN_OUT * D);
  cvt(wq, wq16, (long long)L * D * HD);
  cvt(wk, wk16, (long long)L * D * HD);
  cvt(wv, wv16, (long long)L * D * D);
  cvt(wo, wo16, (long long)L * D * D);
  cvt(cls_w1, w1c16, (long long)D * 256);
  pad_w2<<<dim3(96), dim3(256), 0, stream>>>(cls_w2, cls_b2, w2p16, b2p);

  // ---- CNN front-end ----
  {
    long long n1 = (long long)8 * 32 * F0 * T;
    conv1_direct<<<dim3((unsigned)((n1 + 255) / 256)), dim3(256), 0, stream>>>(
        x, conv_w[0], bn_g[0], bn_b[0], bn_m[0], bn_v[0], P, F0, T);
  }
  conv_wmma<<<dim3(F0 * T / 64, 64 / 64, 8), dim3(128), 0, stream>>>(
      P, w2f, bn_g[1], bn_b[1], bn_m[1], bn_v[1], Q, 32, 64, F0, T);
  {
    long long n = (long long)8 * 64 * F1 * T;
    maxpool_f2<<<dim3((unsigned)((n + 255) / 256)), dim3(256), 0, stream>>>(Q, P, 64, F0, F1, T);
  }
  conv_wmma<<<dim3(F1 * T / 64, 128 / 64, 8), dim3(128), 0, stream>>>(
      P, w3f, bn_g[2], bn_b[2], bn_m[2], bn_v[2], Q, 64, 128, F1, T);
  {
    long long n = (long long)8 * 128 * F2 * T;
    maxpool_f2<<<dim3((unsigned)((n + 255) / 256)), dim3(256), 0, stream>>>(Q, P, 128, F1, F2, T);
  }
  conv_wmma<<<dim3(F2 * T / 64, 256 / 64, 8), dim3(128), 0, stream>>>(
      P, w4f, bn_g[3], bn_b[3], bn_m[3], bn_v[3], Q, 128, 256, F2, T);
  {
    long long n = (long long)8 * 256 * F3 * T;
    maxpool_f2<<<dim3((unsigned)((n + 255) / 256)), dim3(256), 0, stream>>>(Q, P, 256, F2, F3, T);
  }
  {
    long long n = (long long)8 * T * CNN_OUT;
    transpose_cft<<<dim3((unsigned)((n + 255) / 256)), dim3(256), 0, stream>>>(P, Q, 256, F3, T);
  }

  // ---- projection + LN0 + positional encoding ----
  gemm_nn_wmma<<<dim3(D / 64, M / 64, 1), dim3(128), 0, stream>>>(
      Q, projw16, proj_b, nullptr, h, nullptr,
      D, CNN_OUT, CNN_OUT, D, D, 0, 0, 0, 1.0f, 0);
  layernorm<<<dim3(M), dim3(256), 0, stream>>>(h, ln0_g, ln0_b, h, nullptr, D, 1);

  // ---- xLSTM-attention layers ----
  const float scale = 0.125f;  // 1/sqrt(64)
  for (int l = 0; l < L; ++l) {
    layernorm<<<dim3(M), dim3(256), 0, stream>>>(h, lng + l * D, lnb + l * D,
                                                 nullptr, z16, D, 0);
    gemm_nn_wmma<<<dim3(1, M / 64, 1), dim3(128), 0, stream>>>(
        z16, wq16 + (long long)l * D * HD, bq + l * HD, nullptr, nullptr, q16,
        HD, D, D, HD, HD, 0, 0, 0, 1.0f, 0);
    gemm_nn_wmma<<<dim3(1, M / 64, 1), dim3(128), 0, stream>>>(
        z16, wk16 + (long long)l * D * HD, bk + l * HD, nullptr, nullptr, k16,
        HD, D, D, HD, HD, 0, 0, 0, 1.0f, 0);
    gemm_nn_wmma<<<dim3(D / 64, M / 64, 1), dim3(128), 0, stream>>>(
        z16, wv16 + (long long)l * D * D, bv + l * D, nullptr, nullptr, v16,
        D, D, D, D, D, 0, 0, 0, 1.0f, 0);
    gate_kernel<<<dim3(M), dim3(128), 0, stream>>>(z16, wi + l * D, bi + l, gate, D);

    // attn = scale * q k^T  (per batch)
    gemm_tn_wmma<<<dim3(T / 64, T / 16, B), dim3(128), 0, stream>>>(
        q16, k16, attn,
        T, HD, HD, HD, T,
        (long long)T * HD, (long long)T * HD, (long long)T * T, scale);
    softmax_gate<<<dim3(T, B), dim3(256), 0, stream>>>(attn, gate, gated, T);

    // o = gated @ v (per batch)
    gemm_nn_wmma<<<dim3(D / 64, T / 64, B), dim3(128), 0, stream>>>(
        gated, v16, nullptr, nullptr, nullptr, o16,
        D, T, T, D, D,
        (long long)T * T, (long long)T * D, (long long)T * D, 1.0f, 0);

    // h = h + o @ wo + bo
    gemm_nn_wmma<<<dim3(D / 64, M / 64, 1), dim3(128), 0, stream>>>(
        o16, wo16 + (long long)l * D * D, bo + l * D, h, h, nullptr,
        D, D, D, D, D, 0, 0, 0, 1.0f, 0);
  }

  // ---- classifier head ----
  layernorm<<<dim3(M), dim3(256), 0, stream>>>(h, cls_lng, cls_lnb, nullptr, hcls16, D, 0);
  gemm_nn_wmma<<<dim3(256 / 64, M / 64, 1), dim3(128), 0, stream>>>(
      hcls16, w1c16, cls_b1, nullptr, nullptr, c1_16,
      256, D, D, 256, 256, 0, 0, 0, 1.0f, 2 /*ELU*/);
  gemm_nn_wmma<<<dim3(2, M / 64, 1), dim3(128), 0, stream>>>(
      c1_16, w2p16, b2p, nullptr, outp, nullptr,
      96, 256, 256, 96, 96, 0, 0, 0, 1.0f, 0);
  {
    long long n = (long long)8 * 88 * T;
    write_out<<<dim3((unsigned)((n + 255) / 256)), dim3(256), 0, stream>>>(
        outp, (float*)d_out, T, 88, 96);
  }
}
